// MaxSim_37752762531997
// MI455X (gfx1250) — matrix operand order repaired and checked by reading, not compiled, not run on hardware
//
#include <hip/hip_runtime.h>

// MaxSim contrastive loss for MI455X (gfx1250, wave32).
//
// Roofline: neg einsum = 34.4 GFLOP (f16 WMMA, fp32 accum). Working set
// (17 MB f16) lives in the 192 MB L2, so the limiter is L2->VGPR traffic.
// Round-1 design moved ~2.1 GB from L2 (keys re-read per M-tile, per pair).
// This round groups G=4 query batches per negative block so each B-fragment
// feeds 8 WMMAs (4 batches x 2 M-tiles): key traffic ~0.27 GB, q tiles staged
// in LDS once per block via the Tensor Data Mover (tensor_load_to_lds +
// s_wait_tensorcnt) and re-read through DS loads.
//
// Pipeline:
//  1) norm_f16_kernel  : L2-normalize rows (fp32 -> f16 in d_ws).
//  2) maxsim_pos_kernel: B blocks, one (b, pos_key[b]) pair each.
//  3) maxsim_neg_kernel: (B/4)*B blocks; block = key set n x batches
//     [4g, 4g+3]; mask = pos_mask[n] (faithful to reference).
//  4) loss_kernel      : 128x129 log-softmax CE, mean -> d_out[0].

#define B_       128
#define LQ       32
#define LK       256
#define DD       128
#define G_       4
#define TEMP_INV 20.0f            // 1 / 0.05
#define NEG_INF  (-__builtin_inff())

typedef __attribute__((ext_vector_type(16))) _Float16 v16h;
typedef __attribute__((ext_vector_type(8)))  _Float16 v8h;
typedef __attribute__((ext_vector_type(4)))  _Float16 v4h;
typedef __attribute__((ext_vector_type(8)))  float    v8f;
typedef __attribute__((ext_vector_type(4)))  float    v4f;
typedef __attribute__((ext_vector_type(4)))  unsigned int v4u;
typedef __attribute__((ext_vector_type(8)))  int      v8i;
typedef __attribute__((ext_vector_type(4)))  int      v4i;

// ---------------------------------------------------------------------------
// Kernel 1: L2-normalize rows of length D=128, emit f16. One wave per row.
// ---------------------------------------------------------------------------
__global__ __launch_bounds__(128)
void norm_f16_kernel(const float* __restrict__ src, _Float16* __restrict__ dst,
                     int nrows) {
  const int wave = threadIdx.x >> 5;
  const int lane = threadIdx.x & 31;
  const int row  = blockIdx.x * 4 + wave;
  if (row >= nrows) return;

  v4f x = *(const v4f*)(src + (size_t)row * DD + lane * 4);
  float s = x[0]*x[0] + x[1]*x[1] + x[2]*x[2] + x[3]*x[3];
#pragma unroll
  for (int off = 16; off >= 1; off >>= 1) s += __shfl_xor(s, off, 32);
  const float inv = 1.0f / fmaxf(sqrtf(s), 1e-12f);

  v4h h;
  h[0] = (_Float16)(x[0] * inv);
  h[1] = (_Float16)(x[1] * inv);
  h[2] = (_Float16)(x[2] * inv);
  h[3] = (_Float16)(x[3] * inv);
  *(v4h*)(dst + (size_t)row * DD + lane * 4) = h;
}

// Fragment layouts used here (wave32); K order is contiguous, not the ISA 7.12.2 table:
//  A 16x32 f16: lane l -> row l%16; halves j: K = 16*(l/16)+j (two 16B loads)
//  B 32x16    : lane l -> col l%16; halves j: K = 16*(l/16)+j (two 16B loads)
//  C 16x16 f32: VGPR r, group g=(l/16): element (M = r+8g, N = l%16)
__device__ __forceinline__ v16h pack16(v8h lo, v8h hi) {
  v16h r;
#pragma unroll
  for (int i = 0; i < 8; ++i) { r[i] = lo[i]; r[8 + i] = hi[i]; }
  return r;
}

// ---------------------------------------------------------------------------
// Kernel 2: positive pairs (1/129 of the FLOPs) — one block per b.
// ---------------------------------------------------------------------------
__global__ __launch_bounds__(128)
void maxsim_pos_kernel(const _Float16* __restrict__ qn,
                       const _Float16* __restrict__ pkn,
                       const unsigned char* __restrict__ qmask,
                       const unsigned char* __restrict__ pmask,
                       float* __restrict__ scores) {
  __shared__ float s_rowmax[4][LQ];

  const int b    = blockIdx.x;
  const int lane = threadIdx.x & 31;
  const int wave = threadIdx.x >> 5;
  const int half = lane >> 4;
  const int l16  = lane & 15;

  const _Float16* keys  = pkn + (size_t)b * LK * DD;
  const unsigned char* km = pmask + (size_t)b * LK;
  const _Float16* qbase = qn + (size_t)b * LQ * DD;

#pragma unroll
  for (int mt = 0; mt < 2; ++mt) {
    v16h afrag[4];
    const _Float16* arow = qbase + (size_t)(mt * 16 + l16) * DD;
#pragma unroll
    for (int kc = 0; kc < 4; ++kc) {
      const int d0 = kc * 32 + 16 * half;
      afrag[kc] = pack16(*(const v8h*)(arow + d0), *(const v8h*)(arow + d0 + 8));
    }
    v8f vmax;
#pragma unroll
    for (int r = 0; r < 8; ++r) vmax[r] = NEG_INF;

#pragma unroll
    for (int t = 0; t < 4; ++t) {
      const int nt = wave + 4 * t;
      const _Float16* krow = keys + (size_t)(nt * 16 + l16) * DD;
      v8f c = {};
#pragma unroll
      for (int kc = 0; kc < 4; ++kc) {
        const int d0 = kc * 32 + 16 * half;
        v16h bf = pack16(*(const v8h*)(krow + d0), *(const v8h*)(krow + d0 + 8));
        c = __builtin_amdgcn_wmma_f32_16x16x32_f16(
                false, afrag[kc], false, bf, (short)0, c, false, false);
      }
      const bool kv = km[nt * 16 + l16] != 0;
#pragma unroll
      for (int r = 0; r < 8; ++r) vmax[r] = fmaxf(vmax[r], kv ? c[r] : NEG_INF);
    }
#pragma unroll
    for (int r = 0; r < 8; ++r) {
      float v = vmax[r];
#pragma unroll
      for (int off = 1; off <= 8; off <<= 1) v = fmaxf(v, __shfl_xor(v, off, 32));
      vmax[r] = v;
    }
    if (l16 == 0) {
      const int base = mt * 16 + half * 8;
#pragma unroll
      for (int r = 0; r < 8; ++r) s_rowmax[wave][base + r] = vmax[r];
    }
  }
  __syncthreads();

  if (threadIdx.x < LQ) {
    const int row = threadIdx.x;
    float m = fmaxf(fmaxf(s_rowmax[0][row], s_rowmax[1][row]),
                    fmaxf(s_rowmax[2][row], s_rowmax[3][row]));
    float v = (qmask[(size_t)b * LQ + row] != 0) ? m : 0.0f;
#pragma unroll
    for (int off = 16; off >= 1; off >>= 1) v += __shfl_xor(v, off, 32);
    if (row == 0) scores[b] = v;
  }
}

// ---------------------------------------------------------------------------
// Kernel 3: negative pairs, G_=4 batches per block. Key fragments loaded from
// L2 once per block and reused across 8 M-tiles; the group's q tiles (32 KB)
// are staged into LDS by one Tensor Data Mover op (wave 0) and read via DS.
// ---------------------------------------------------------------------------
__global__ __launch_bounds__(128)
void maxsim_neg_kernel(const _Float16* __restrict__ qn,
                       const _Float16* __restrict__ nkn,
                       const unsigned char* __restrict__ qmask,
                       const unsigned char* __restrict__ pmask,
                       float* __restrict__ scores) {
  __shared__ _Float16 s_q[G_ * LQ * DD];      // 32 KB
  __shared__ float s_rowmax[4][G_][LQ];       // 2 KB

  const int lane = threadIdx.x & 31;
  const int wave = threadIdx.x >> 5;
  const int half = lane >> 4;
  const int l16  = lane & 15;

  const int n  = blockIdx.x % B_;             // key set (neg_key[n], pos_mask[n])
  const int gb = (blockIdx.x / B_) * G_;      // first query batch of the group

  // --- TDM: stage qn[gb*LQ .. gb*LQ+127][0..127] (contiguous 32 KB) into LDS.
  // D# per ISA 8.3/8.4: 1 "row" of 16384 2-byte elements.
  if (threadIdx.x < 32) {
    const unsigned lds_off = (unsigned)(size_t)(&s_q[0]);
    const unsigned long long ga =
        (unsigned long long)(size_t)(qn + (size_t)gb * LQ * DD);
    v4u d0;
    d0[0] = 1u;                                              // count=1
    d0[1] = lds_off;                                         // lds_addr
    d0[2] = (unsigned)(ga & 0xffffffffu);                    // global_addr lo
    d0[3] = (unsigned)((ga >> 32) & 0x01ffffffu) | (2u << 30); // addr hi | type=2
    v8i d1;
    d1[0] = 0x10000;                 // data_size=1 (2 bytes/elem)
    d1[1] = (int)(16384u << 16);     // tensor_dim0 = 16384 (bits 79:48)
    d1[2] = 0x10000;                 // tensor_dim0 hi=0 | tensor_dim1 = 1
    d1[3] = (int)(16384u << 16);     // tensor_dim1 hi=0 | tile_dim0 = 16384
    d1[4] = 1;                       // tile_dim1 = 1 | tile_dim2 = 0
    d1[5] = 16384;                   // tensor_dim0_stride lo
    d1[6] = (int)(16384u << 16);     // stride0 hi=0 | tensor_dim1_stride lo
    d1[7] = 0;                       // tensor_dim1_stride hi
    v4i zz4 = (v4i){0, 0, 0, 0};
    v8i zz8 = (v8i){0, 0, 0, 0, 0, 0, 0, 0};
    __builtin_amdgcn_tensor_load_to_lds(d0, d1, zz4, zz4, zz8, 0);
    __builtin_amdgcn_s_wait_tensorcnt(0);
  }
  __syncthreads();

  const _Float16* keys = nkn + (size_t)n * LK * DD;
  const unsigned char* km = pmask + (size_t)n * LK;

  // Row-max accumulators: tile = g*2 + mt covers q rows [tile*16, tile*16+16).
  v8f vmax[2 * G_];
#pragma unroll
  for (int t = 0; t < 2 * G_; ++t)
#pragma unroll
    for (int r = 0; r < 8; ++r) vmax[t][r] = NEG_INF;

#pragma unroll
  for (int t = 0; t < 4; ++t) {
    const int nt = wave + 4 * t;                  // this wave's N-tile
    const _Float16* krow = keys + (size_t)(nt * 16 + l16) * DD;
    v16h bfrag[4];
#pragma unroll
    for (int kc = 0; kc < 4; ++kc) {
      const int d0 = kc * 32 + 16 * half;
      bfrag[kc] = pack16(*(const v8h*)(krow + d0), *(const v8h*)(krow + d0 + 8));
    }
    const bool kv = km[nt * 16 + l16] != 0;

#pragma unroll
    for (int tile = 0; tile < 2 * G_; ++tile) {   // 8 M-tiles reuse bfrag
      const _Float16* arow = s_q + (size_t)(tile * 16 + l16) * DD;
      v8f c = {};
#pragma unroll
      for (int kc = 0; kc < 4; ++kc) {
        const int d0 = kc * 32 + 16 * half;
        v16h af = pack16(*(const v8h*)(arow + d0), *(const v8h*)(arow + d0 + 8));
        c = __builtin_amdgcn_wmma_f32_16x16x32_f16(
                false, af, false, bfrag[kc], (short)0, c, false, false);
      }
#pragma unroll
      for (int r = 0; r < 8; ++r)
        vmax[tile][r] = fmaxf(vmax[tile][r], kv ? c[r] : NEG_INF);
    }
  }

  // Reduce each tile's row-max over the 16 lanes of each half-wave.
#pragma unroll
  for (int tile = 0; tile < 2 * G_; ++tile) {
#pragma unroll
    for (int r = 0; r < 8; ++r) {
      float v = vmax[tile][r];
#pragma unroll
      for (int off = 1; off <= 8; off <<= 1) v = fmaxf(v, __shfl_xor(v, off, 32));
      vmax[tile][r] = v;
    }
    if (l16 == 0) {
      const int g = tile >> 1, mt = tile & 1;
      const int base = mt * 16 + half * 8;        // C rows: M = r + 8*half
#pragma unroll
      for (int r = 0; r < 8; ++r) s_rowmax[wave][g][base + r] = vmax[tile][r];
    }
  }
  __syncthreads();

  // Wave g: cross-wave max over N-subsets, gate by query mask, sum rows.
  {
    const int g = wave;
    const int row = lane;
    float m = fmaxf(fmaxf(s_rowmax[0][g][row], s_rowmax[1][g][row]),
                    fmaxf(s_rowmax[2][g][row], s_rowmax[3][g][row]));
    const int b = gb + g;
    float v = (qmask[(size_t)b * LQ + row] != 0) ? m : 0.0f;
#pragma unroll
    for (int off = 16; off >= 1; off >>= 1) v += __shfl_xor(v, off, 32);
    if (lane == 0) scores[B_ + b * B_ + n] = v;
  }
}

// ---------------------------------------------------------------------------
// Kernel 4: loss = mean_b ( logsumexp(logits_b) - logits_b[0] ).
// ---------------------------------------------------------------------------
__global__ __launch_bounds__(256)
void loss_kernel(const float* __restrict__ scores, float* __restrict__ out) {
  __shared__ float s_acc[8];
  const int lane = threadIdx.x & 31;
  const int wave = threadIdx.x >> 5;

  float acc = 0.0f;
  for (int b = wave; b < B_; b += 8) {
    float lmax = NEG_INF;
    for (int i = lane; i <= B_; i += 32) {
      const float v = (i == 0 ? scores[b] : scores[B_ + b * B_ + (i - 1)]) * TEMP_INV;
      lmax = fmaxf(lmax, v);
    }
#pragma unroll
    for (int off = 16; off >= 1; off >>= 1) lmax = fmaxf(lmax, __shfl_xor(lmax, off, 32));

    float s = 0.0f;
    for (int i = lane; i <= B_; i += 32) {
      const float v = (i == 0 ? scores[b] : scores[B_ + b * B_ + (i - 1)]) * TEMP_INV;
      s += __expf(v - lmax);
    }
#pragma unroll
    for (int off = 16; off >= 1; off >>= 1) s += __shfl_xor(s, off, 32);

    acc += (lmax + __logf(s)) - scores[b] * TEMP_INV;
  }
  if (lane == 0) s_acc[wave] = acc;
  __syncthreads();
  if (threadIdx.x == 0) {
    float t = 0.0f;
#pragma unroll
    for (int w = 0; w < 8; ++w) t += s_acc[w];
    out[0] = t / (float)B_;
  }
}

// ---------------------------------------------------------------------------
extern "C" void kernel_launch(void* const* d_in, const int* in_sizes, int n_in,
                              void* d_out, int out_size, void* d_ws, size_t ws_size,
                              hipStream_t stream) {
  (void)in_sizes; (void)n_in; (void)out_size; (void)ws_size;

  const float* query   = (const float*)d_in[0];
  const float* pos_key = (const float*)d_in[1];
  const float* neg_key = (const float*)d_in[2];
  const unsigned char* qmask = (const unsigned char*)d_in[3];  // bool, 1B/elem
  const unsigned char* pmask = (const unsigned char*)d_in[4];  // bool, 1B/elem
  // d_in[5] (neg_mask) is unused by the reference.

  char* ws = (char*)d_ws;
  size_t off = 0;
  _Float16* qn  = (_Float16*)(ws + off); off += (size_t)B_ * LQ * DD * 2;  // 1 MB
  _Float16* pkn = (_Float16*)(ws + off); off += (size_t)B_ * LK * DD * 2;  // 8 MB
  _Float16* nkn = (_Float16*)(ws + off); off += (size_t)B_ * LK * DD * 2;  // 8 MB
  float* scores = (float*)(ws + off);    off += (size_t)B_ * (B_ + 1) * 4; // 64.5 KB

  norm_f16_kernel<<<(B_ * LQ) / 4, 128, 0, stream>>>(query,   qn,  B_ * LQ);
  norm_f16_kernel<<<(B_ * LK) / 4, 128, 0, stream>>>(pos_key, pkn, B_ * LK);
  norm_f16_kernel<<<(B_ * LK) / 4, 128, 0, stream>>>(neg_key, nkn, B_ * LK);

  maxsim_pos_kernel<<<B_, 128, 0, stream>>>(qn, pkn, qmask, pmask, scores);
  maxsim_neg_kernel<<<(B_ / G_) * B_, 128, 0, stream>>>(qn, nkn, qmask, pmask, scores);

  loss_kernel<<<1, 256, 0, stream>>>(scores, (float*)d_out);
}
